// CustomAARDecoder_66606352826848
// MI455X (gfx1250) — compile-verified, matrix operand order checked by reading, not run on hardware
//
#include <hip/hip_runtime.h>
#include <hip/hip_bf16.h>

// ---------------------------------------------------------------------------
// CDNA5 (gfx1250) bf16 WMMA implementation of the 6-layer cross-attention
// decoder + LM head.  All GEMMs use v_wmma_f32_16x16x32_bf16 (wave32),
// double-buffered LDS, and GLOBAL_LOAD_ASYNC_TO_LDS_B128 + s_wait_asynccnt
// for the staging pipeline (falls back to reg staging if not exposed).
// ---------------------------------------------------------------------------

#ifndef __has_builtin
#define __has_builtin(x) 0
#endif

#if __has_builtin(__builtin_amdgcn_global_load_async_to_lds_b128)
#define USE_ASYNC_LDS 1
#else
#define USE_ASYNC_LDS 0
#endif

typedef __attribute__((ext_vector_type(16))) __bf16 v16bf;
typedef __attribute__((ext_vector_type(8)))  __bf16 v8bf;
typedef __attribute__((ext_vector_type(8)))  float  v8f;
typedef unsigned int u32x4 __attribute__((ext_vector_type(4)));
typedef int gv4i __attribute__((vector_size(4 * sizeof(int))));

#define TB   256          // threads per block (8 wave32)
#define BM   128          // block tile M
#define BN   128          // block tile N
#define BK   32           // k chunk
#define LDSK 40           // padded LDS row stride in halves (80B, 16B-aligned)
#define BUFH ((BM + BN) * LDSK)   // halves per LDS buffer (A tile + B tile)

// Output epilogue modes
#define OM_BF16        0  // bf16, C[row*ldc+col] = bf16(acc*scale)
#define OM_BF16_T      1  // bf16 transposed, C[col*ldc+row]
#define OM_F32_BIAS    2  // f32, C = acc + bias[col]  (bias may be null)
#define OM_BF16_BIAS   3  // bf16, C = bf16(acc + bias[col])

__device__ __forceinline__ v16bf cat16(v8bf lo, v8bf hi) {
    return __builtin_shufflevector(lo, hi, 0,1,2,3,4,5,6,7,8,9,10,11,12,13,14,15);
}

#if USE_ASYNC_LDS
__device__ __forceinline__ void async_cp16(const __bf16* g, __bf16* l) {
    __builtin_amdgcn_global_load_async_to_lds_b128(
        (__attribute__((address_space(1))) gv4i*)g,
        (__attribute__((address_space(3))) gv4i*)l,
        0, 0);
}
__device__ __forceinline__ void async_wait0() {
#if __has_builtin(__builtin_amdgcn_s_wait_asynccnt)
    __builtin_amdgcn_s_wait_asynccnt(0);
#else
    asm volatile("s_wait_asynccnt 0x0" ::: "memory");
#endif
}
#endif

// C[M,N] = scale * A[M,K] x B[N,K]^T   (both bf16 row-major, K contiguous)
// batched over blockIdx.z with element strides sA/sB/sC.
// Block 128x128, 8 waves in a 2(M) x 4(N) grid, each wave 64x32 = 4x2 WMMA.
template<int OMODE>
__global__ __launch_bounds__(TB)
void gemm_nt_bf16(const __bf16* __restrict__ A,
                  const __bf16* __restrict__ Bm,
                  void* __restrict__ Cv,
                  const float* __restrict__ bias,
                  int lda, int ldb, int ldc, int K,
                  long long sA, long long sB, long long sC,
                  float scale)
{
    __shared__ __bf16 sm[2][BUFH];    // [buf][A tile | B tile], 40 KB total

    const int t    = threadIdx.x;
    const int lane = t & 31;
    const int wv   = t >> 5;
    const int wm   = wv & 1;          // 2 waves along M (64 rows each)
    const int wn   = wv >> 1;         // 4 waves along N (32 cols each)
    const int bm   = blockIdx.x * BM;
    const int bn   = blockIdx.y * BN;
    const int z    = blockIdx.z;

    A  += (long long)z * sA;
    Bm += (long long)z * sB;

    const int rlane   = lane & 15;
    const int halfSel = lane >> 4;    // 0: K 0-7/16-23, 1: K 8-15/24-31
    const int kb      = halfSel * 8;

    // staging: 4 threads per row, 8 halves (16B) each; rows gr and gr+64
    const int gr  = t >> 2;           // 0..63
    const int gc8 = (t & 3) * 8;      // half offset within k-chunk
    const long long ga0 = (long long)(bm + gr)      * lda + gc8;
    const long long ga1 = (long long)(bm + gr + 64) * lda + gc8;
    const long long gb0 = (long long)(bn + gr)      * ldb + gc8;
    const long long gb1 = (long long)(bn + gr + 64) * ldb + gc8;
    const int la0 = gr * LDSK + gc8;
    const int la1 = (gr + 64) * LDSK + gc8;
    const int lb0 = BM * LDSK + la0;
    const int lb1 = BM * LDSK + la1;

    v8f acc[4][2] = {};

    // ---- prologue: stage chunk 0 into buffer 0 -------------------------
#if USE_ASYNC_LDS
    async_cp16(&A[ga0],  &sm[0][la0]);
    async_cp16(&A[ga1],  &sm[0][la1]);
    async_cp16(&Bm[gb0], &sm[0][lb0]);
    async_cp16(&Bm[gb1], &sm[0][lb1]);
    async_wait0();
#else
    *(u32x4*)&sm[0][la0] = *(const u32x4*)&A[ga0];
    *(u32x4*)&sm[0][la1] = *(const u32x4*)&A[ga1];
    *(u32x4*)&sm[0][lb0] = *(const u32x4*)&Bm[gb0];
    *(u32x4*)&sm[0][lb1] = *(const u32x4*)&Bm[gb1];
#endif
    __syncthreads();

    int cur = 0;
    for (int kc = 0; kc < K; kc += BK) {
        const int  nxt  = cur ^ 1;
        const bool more = (kc + BK) < K;
        const int  kn   = kc + BK;

#if USE_ASYNC_LDS
        // kick off next chunk's DMA while this chunk is multiplied
        if (more) {
            async_cp16(&A[ga0 + kn],  &sm[nxt][la0]);
            async_cp16(&A[ga1 + kn],  &sm[nxt][la1]);
            async_cp16(&Bm[gb0 + kn], &sm[nxt][lb0]);
            async_cp16(&Bm[gb1 + kn], &sm[nxt][lb1]);
        }
#else
        u32x4 ra0, ra1, rb0, rb1;
        if (more) {
            ra0 = *(const u32x4*)&A[ga0 + kn];
            ra1 = *(const u32x4*)&A[ga1 + kn];
            rb0 = *(const u32x4*)&Bm[gb0 + kn];
            rb1 = *(const u32x4*)&Bm[gb1 + kn];
        }
#endif

        // ---- fragments per ISA 16-bit A/B VGPR layout + 8 WMMAs --------
        const __bf16* lA = &sm[cur][0];
        const __bf16* lB = &sm[cur][BM * LDSK];

        v16bf af[4], bf[2];
#pragma unroll
        for (int tm = 0; tm < 4; ++tm) {
            const int row = (wm * 64 + tm * 16 + rlane) * LDSK;
            af[tm] = cat16(*(const v8bf*)&lA[row + kb],
                           *(const v8bf*)&lA[row + kb + 16]);
        }
#pragma unroll
        for (int tn = 0; tn < 2; ++tn) {
            const int row = (wn * 32 + tn * 16 + rlane) * LDSK;
            bf[tn] = cat16(*(const v8bf*)&lB[row + kb],
                           *(const v8bf*)&lB[row + kb + 16]);
        }
#pragma unroll
        for (int tm = 0; tm < 4; ++tm)
#pragma unroll
            for (int tn = 0; tn < 2; ++tn)
                acc[tm][tn] = __builtin_amdgcn_wmma_f32_16x16x32_bf16(
                    false, af[tm], false, bf[tn],
                    (short)0, acc[tm][tn], false, false);

#if USE_ASYNC_LDS
        if (more) async_wait0();
#else
        if (more) {
            *(u32x4*)&sm[nxt][la0] = ra0;
            *(u32x4*)&sm[nxt][la1] = ra1;
            *(u32x4*)&sm[nxt][lb0] = rb0;
            *(u32x4*)&sm[nxt][lb1] = rb1;
        }
#endif
        __syncthreads();
        cur = nxt;
    }

    // ---- epilogue ------------------------------------------------------
    const long long cz = (long long)z * sC;
#pragma unroll
    for (int tm = 0; tm < 4; ++tm)
#pragma unroll
        for (int tn = 0; tn < 2; ++tn)
#pragma unroll
            for (int r = 0; r < 8; ++r) {
                const int row = bm + wm * 64 + tm * 16 + r + halfSel * 8;
                const int col = bn + wn * 32 + tn * 16 + rlane;
                float v = acc[tm][tn][r] * scale;
                if (OMODE == OM_BF16) {
                    ((__bf16*)Cv)[cz + (long long)row * ldc + col] = (__bf16)v;
                } else if (OMODE == OM_BF16_T) {
                    ((__bf16*)Cv)[cz + (long long)col * ldc + row] = (__bf16)v;
                } else if (OMODE == OM_F32_BIAS) {
                    if (bias) v += bias[col];
                    ((float*)Cv)[cz + (long long)row * ldc + col] = v;
                } else { // OM_BF16_BIAS
                    v += bias[col];
                    ((__bf16*)Cv)[cz + (long long)row * ldc + col] = (__bf16)v;
                }
            }
}

// ---------------------------------------------------------------------------
// support kernels
// ---------------------------------------------------------------------------

__global__ void f32_to_bf16_kernel(const float* __restrict__ in,
                                   __bf16* __restrict__ out, long long n)
{
    long long i = (long long)blockIdx.x * blockDim.x + threadIdx.x;
    const long long stride = (long long)gridDim.x * blockDim.x;
    for (; i < n; i += stride) out[i] = (__bf16)in[i];
}

__global__ void embed_gather_kernel(const int* __restrict__ idx,
                                    const float* __restrict__ embed,
                                    __bf16* __restrict__ x, int E)
{
    const long long row = blockIdx.x;
    const long long tok = idx[row];
    const float* e = embed + tok * E;
    __bf16* xr = x + row * E;
    for (int i = threadIdx.x; i < E; i += blockDim.x) xr[i] = (__bf16)e[i];
}

__global__ __launch_bounds__(256)
void softmax_rows_kernel(const float* __restrict__ S,
                         __bf16* __restrict__ A, int n)
{
    __shared__ float red[256];
    const long long row = blockIdx.x;
    const float* s = S + row * n;
    __bf16* a = A + row * n;
    const int t = threadIdx.x;

    float m = -3.402823466e+38f;
    for (int i = t; i < n; i += 256) m = fmaxf(m, s[i]);
    red[t] = m; __syncthreads();
    for (int o = 128; o > 0; o >>= 1) {
        if (t < o) red[t] = fmaxf(red[t], red[t + o]);
        __syncthreads();
    }
    m = red[0]; __syncthreads();

    float sum = 0.f;
    for (int i = t; i < n; i += 256) sum += __expf(s[i] - m);
    red[t] = sum; __syncthreads();
    for (int o = 128; o > 0; o >>= 1) {
        if (t < o) red[t] += red[t + o];
        __syncthreads();
    }
    const float inv = 1.f / red[0];
    for (int i = t; i < n; i += 256) a[i] = (__bf16)(__expf(s[i] - m) * inv);
}

// ---------------------------------------------------------------------------
// orchestration
// ---------------------------------------------------------------------------

extern "C" void kernel_launch(void* const* d_in, const int* in_sizes, int n_in,
                              void* d_out, int out_size, void* d_ws, size_t ws_size,
                              hipStream_t stream)
{
    constexpr int B = 8, SQ = 1024, SKV = 1024, E = 1024, V = 32000, L = 6;
    constexpr int M  = B * SQ;         // 8192 query rows
    constexpr int MK = B * SKV;        // 8192 kv rows
    const float SCALE = 0.125f;        // (E/H)^-0.5 = 64^-0.5

    const int*   input_text = (const int*)  d_in[0];  // B*SQ
    const float* target     = (const float*)d_in[1];  // B*SKV*E
    const float* embed      = (const float*)d_in[2];  // V*E
    const float* in_proj_w  = (const float*)d_in[3];  // L*3E*E
    const float* out_w      = (const float*)d_in[4];  // L*E*E
    const float* out_b      = (const float*)d_in[5];  // L*E
    const float* lin_w      = (const float*)d_in[6];  // V*E
    const float* lin_b      = (const float*)d_in[7];  // V
    float*       logits     = (float*)d_out;          // B*SQ*V

    // ---- workspace carve ----------------------------------------------
    char* w = (char*)d_ws;
    auto carve = [&](long long bytes) -> void* {
        void* p = (void*)w;
        w += (bytes + 255) & ~255LL;
        return p;
    };
    __bf16* x0      = (__bf16*)carve((long long)M * E * 2);
    __bf16* x1      = (__bf16*)carve((long long)M * E * 2);
    __bf16* tgt     = (__bf16*)carve((long long)MK * E * 2);
    __bf16* qb      = (__bf16*)carve((long long)M * E * 2);
    __bf16* kb      = (__bf16*)carve((long long)MK * E * 2);
    __bf16* vT      = (__bf16*)carve((long long)B * E * SKV * 2);
    float*  scores  = (float*) carve((long long)B * SQ * SKV * 4);
    __bf16* attn    = (__bf16*)carve((long long)B * SQ * SKV * 2);
    __bf16* ob      = (__bf16*)carve((long long)M * E * 2);
    __bf16* wproj   = (__bf16*)carve((long long)L * 3 * E * E * 2);
    __bf16* woutb   = (__bf16*)carve((long long)L * E * E * 2);
    __bf16* wlin    = (__bf16*)carve((long long)V * E * 2);

    // ---- one-time-per-call conversions to bf16 ------------------------
    f32_to_bf16_kernel<<<4096, 256, 0, stream>>>(target, tgt, (long long)MK * E);
    f32_to_bf16_kernel<<<4096, 256, 0, stream>>>(in_proj_w, wproj, (long long)L * 3 * E * E);
    f32_to_bf16_kernel<<<4096, 256, 0, stream>>>(out_w, woutb, (long long)L * E * E);
    f32_to_bf16_kernel<<<8192, 256, 0, stream>>>(lin_w, wlin, (long long)V * E);
    embed_gather_kernel<<<M, 256, 0, stream>>>(input_text, embed, x0, E);

    __bf16* x  = x0;
    __bf16* xn = x1;

    for (int l = 0; l < L; ++l) {
        const __bf16* Wq = wproj + (long long)l * 3 * E * E;
        const __bf16* Wk = Wq + (long long)E * E;
        const __bf16* Wv = Wq + 2LL * E * E;

        // q = (x . Wq^T) * SCALE           [8192 x 1024]
        gemm_nt_bf16<OM_BF16><<<dim3(M / BM, E / BN, 1), TB, 0, stream>>>(
            x, Wq, qb, nullptr, E, E, E, E, 0, 0, 0, SCALE);
        // k = target . Wk^T                [8192 x 1024]
        gemm_nt_bf16<OM_BF16><<<dim3(MK / BM, E / BN, 1), TB, 0, stream>>>(
            tgt, Wk, kb, nullptr, E, E, E, E, 0, 0, 0, 1.0f);
        // vT[b] = (target_b . Wv^T)^T      [E x SKV] per batch
        gemm_nt_bf16<OM_BF16_T><<<dim3(SKV / BM, E / BN, B), TB, 0, stream>>>(
            tgt, Wv, vT, nullptr, E, E, SKV, E,
            (long long)SKV * E, 0, (long long)E * SKV, 1.0f);
        // scores[b] = q_b . k_b^T          [SQ x SKV] per batch, f32
        gemm_nt_bf16<OM_F32_BIAS><<<dim3(SQ / BM, SKV / BN, B), TB, 0, stream>>>(
            qb, kb, scores, nullptr, E, E, SKV, E,
            (long long)SQ * E, (long long)SKV * E, (long long)SQ * SKV, 1.0f);
        // attn = softmax(scores)
        softmax_rows_kernel<<<B * SQ, 256, 0, stream>>>(scores, attn, SKV);
        // o[b] = attn_b . vT_b^T           [SQ x E] per batch
        gemm_nt_bf16<OM_BF16><<<dim3(SQ / BM, E / BN, B), TB, 0, stream>>>(
            attn, vT, ob, nullptr, SKV, SKV, E, SKV,
            (long long)SQ * SKV, (long long)E * SKV, (long long)SQ * E, 1.0f);
        // x = o . Wout^T + out_b           [8192 x 1024]
        gemm_nt_bf16<OM_BF16_BIAS><<<dim3(M / BM, E / BN, 1), TB, 0, stream>>>(
            ob, woutb + (long long)l * E * E, xn, out_b + (long long)l * E,
            E, E, E, E, 0, 0, 0, 1.0f);
        __bf16* tmp = x; x = xn; xn = tmp;
    }

    // logits = x . lin_w^T + lin_b         [8192 x 32000] f32
    gemm_nt_bf16<OM_F32_BIAS><<<dim3(M / BM, V / BN, 1), TB, 0, stream>>>(
        x, wlin, logits, lin_b, E, E, V, E, 0, 0, 0, 1.0f);
}
